// SimCLR_Loss_28544352649834
// MI455X (gfx1250) — matrix operand order repaired and checked by reading, not compiled, not run on hardware
//
#include <hip/hip_runtime.h>
#include <cstddef>

typedef __attribute__((ext_vector_type(16))) _Float16 h16;
typedef __attribute__((ext_vector_type(8)))  _Float16 h8;
typedef __attribute__((ext_vector_type(8)))  float    f8;
typedef __attribute__((ext_vector_type(4)))  float    f4v;

#define N_ROWS     4096
#define TWO_N      8192
#define DK         256
#define ROWS_PER_BLK 128     // 8 waves x 16-row strips
#define COL_TILE   128       // columns staged in LDS per buffer (64 KB)
#define COL_SPLITS 4         // deterministic partial row-sums (no atomics)
#define INV_T      (1.0f / 0.07f)
#define LDS_TILE_HALVES (COL_TILE * DK)   // 32768 halves = 64 KB per buffer

// --------------------------- async copy support ----------------------------
#if __has_builtin(__builtin_amdgcn_global_load_async_to_lds_b128) && \
    __has_builtin(__builtin_amdgcn_s_wait_asynccnt)
#define HAVE_ASYNC 1
// Builtin signature (from clang diagnostic): first arg is a pointer to
// int __vector(4) in the global address space, second in LDS address space.
typedef int v4i_vs __attribute__((vector_size(16)));
typedef __attribute__((address_space(1))) v4i_vs gv4i_t;
typedef __attribute__((address_space(3))) v4i_vs lv4i_t;
#else
#define HAVE_ASYNC 0
#endif

// Issue a 64 KB tile copy (128 z-rows) into LDS. 256 threads * 16B * 16 slots.
// Async path: global -> LDS directly (ASYNCcnt), no VGPR bounce.
__device__ __forceinline__ void tile_copy_issue(const _Float16* __restrict__ src,
                                                _Float16* dst, int tid)
{
#if HAVE_ASYNC
    gv4i_t* g = (gv4i_t*)(const void*)src + tid;   // +16 B per lane slot
    lv4i_t* l = (lv4i_t*)(void*)dst       + tid;
    // INST_OFFSET is added to both the LDS and global addresses (ISA 10.x),
    // so one immediate advances both streams by 4 KB per slot.
#define ASY(i) __builtin_amdgcn_global_load_async_to_lds_b128(g, l, (i) * 4096, 0)
    ASY(0);  ASY(1);  ASY(2);  ASY(3);  ASY(4);  ASY(5);  ASY(6);  ASY(7);
    ASY(8);  ASY(9);  ASY(10); ASY(11); ASY(12); ASY(13); ASY(14); ASY(15);
#undef ASY
#else
    const f4v* s4 = (const f4v*)src;
    f4v*       d4 = (f4v*)dst;
    #pragma unroll
    for (int i = 0; i < 16; ++i)
        d4[tid + i * 256] = s4[tid + i * 256];
#endif
}

__device__ __forceinline__ void tile_copy_wait()
{
#if HAVE_ASYNC
    __builtin_amdgcn_s_wait_asynccnt(0);
#endif
}

// ---------------------------------------------------------------------------
// Kernel 1: L2-normalize rows of emb_i/emb_j -> z (f16, [8192 x 256]),
// and compute positive-pair similarities pos[r] in fp32.
// ---------------------------------------------------------------------------
__global__ void __launch_bounds__(256)
simclr_normalize(const float* __restrict__ ei, const float* __restrict__ ej,
                 _Float16* __restrict__ z, float* __restrict__ pos)
{
    const int row = blockIdx.x;          // 0..4095
    const int t   = threadIdx.x;         // 0..255 == D
    const float xi = ei[(size_t)row * DK + t];
    const float xj = ej[(size_t)row * DK + t];

    float si = xi * xi, sj = xj * xj, sij = xi * xj;
    #pragma unroll
    for (int off = 16; off > 0; off >>= 1) {
        si  += __shfl_down(si,  off, 32);
        sj  += __shfl_down(sj,  off, 32);
        sij += __shfl_down(sij, off, 32);
    }
    __shared__ float r0[8], r1[8], r2[8];
    const int wid = t >> 5, lid = t & 31;
    if (lid == 0) { r0[wid] = si; r1[wid] = sj; r2[wid] = sij; }
    __syncthreads();
    float ti = 0.f, tj = 0.f, tij = 0.f;
    #pragma unroll
    for (int w = 0; w < 8; ++w) { ti += r0[w]; tj += r1[w]; tij += r2[w]; }

    float ni = sqrtf(ti); if (ni < 1e-12f) ni = 1e-12f;
    float nj = sqrtf(tj); if (nj < 1e-12f) nj = 1e-12f;

    z[(size_t)row * DK + t]            = (_Float16)(xi / ni);
    z[(size_t)(row + N_ROWS) * DK + t] = (_Float16)(xj / nj);
    if (t == 0) {
        const float p = tij / (ni * nj);   // sim[i, i+N] == sim[i+N, i]
        pos[row]          = p;
        pos[row + N_ROWS] = p;
    }
}

// ---------------------------------------------------------------------------
// Kernel 2: fused  rowsum_part[split][r] = sum_{c in split} exp(sim[r][c]/T)
// (diagonal masked).  sim tile = Zrow * Zcol^T via v_wmma_f32_16x16x32_f16.
// 128 rows per block; A fragments in VGPRs; double-buffered 64 KB column
// tiles streamed global->LDS asynchronously and shared by all 8 waves.
// ---------------------------------------------------------------------------
__global__ void __launch_bounds__(256)
simclr_expsum(const _Float16* __restrict__ z, float* __restrict__ rowsum_part)
{
    extern __shared__ _Float16 lds[];        // 2 buffers * 64 KB

    const int tid   = threadIdx.x;
    const int lane  = tid & 31;
    const int wave  = tid >> 5;                      // 0..7
    const int row0  = blockIdx.x * ROWS_PER_BLK;
    const int split = blockIdx.y;
    const int col_begin = split * (TWO_N / COL_SPLITS);

    const int strip_row0 = row0 + wave * 16;
    const int mrow       = strip_row0 + (lane & 15); // A-frag source row

    // --- Preload the 8 A fragments (K = 0..255) for this wave's 16 rows. ---
    // A 16x32 layout (matches B): lanes 0-15 hold K {kb..kb+15},
    // lanes 16-31 hold K {kb+16..kb+31}.
    h16 A[8];
    {
        const int khalf = (lane >> 4) * 16;
        const _Float16* rp = z + (size_t)mrow * DK;
        #pragma unroll
        for (int kf = 0; kf < 8; ++kf) {
            const int kb = kf * 32 + khalf;
            A[kf] = *(const h16*)(rp + kb);
        }
    }

    float acc[8];
    #pragma unroll
    for (int r = 0; r < 8; ++r) acc[r] = 0.f;

    // D-matrix row for VGPR r of the accumulator: m_base + r
    const int m_base = strip_row0 + (lane >> 4) * 8;
    const int ksel   = (lane >> 4) * 16;             // B-frag K-half select

    _Float16* buf0 = lds;
    _Float16* buf1 = lds + LDS_TILE_HALVES;

    const int n_tiles = (TWO_N / COL_SPLITS) / COL_TILE;

    // Prime the pipeline with tile 0.
    tile_copy_issue(z + (size_t)col_begin * DK, buf0, tid);

    for (int ct = 0; ct < n_tiles; ++ct) {
        const int colt0 = col_begin + ct * COL_TILE;
        _Float16* cur = (ct & 1) ? buf1 : buf0;
        _Float16* nxt = (ct & 1) ? buf0 : buf1;

        tile_copy_wait();        // this wave's async loads for tile ct done
        __syncthreads();         // all waves' loads done; prev compute done

        // Overlap: stream tile ct+1 into the other buffer (its previous
        // consumers all passed the barrier above).
        if (ct + 1 < n_tiles)
            tile_copy_issue(z + (size_t)(colt0 + COL_TILE) * DK, nxt, tid);

        #pragma unroll 1
        for (int sub = 0; sub < COL_TILE / 16; ++sub) {
            // B 32x16 operand, same k order as A: lane n holds 16 contiguous K of
            // column (sub*16 + (lane&15)); K-half chosen by lane>>4.
            const int col_g = colt0 + sub * 16 + (lane & 15);
            const _Float16* cp = cur + ((sub * 16 + (lane & 15)) * DK) + ksel;

            // Preload all 8 B fragments first so the 16 ds_load_b128 issue
            // ahead of the WMMA chain and the waits retire incrementally
            // (avoids an exposed s_wait_dscnt 0 before every WMMA).
            h16 Bf[8];
            #pragma unroll
            for (int kf = 0; kf < 8; ++kf)
                Bf[kf] = *(const h16*)(cp + kf * 32);

            f8 c = {};
            #pragma unroll
            for (int kf = 0; kf < 8; ++kf)
                c = __builtin_amdgcn_wmma_f32_16x16x32_f16(
                        false, A[kf], false, Bf[kf], (short)0, c, false, false);

            #pragma unroll
            for (int r = 0; r < 8; ++r) {
                float e = __expf(c[r] * INV_T);
                if (m_base + r == col_g) e = 0.f;     // drop diagonal term
                acc[r] += e;
            }
        }
    }

    // Reduce across the 16 lanes of each half (cols) -> per-row sums.
    #pragma unroll
    for (int r = 0; r < 8; ++r) {
        float v = acc[r];
        v += __shfl_xor(v, 1, 16);
        v += __shfl_xor(v, 2, 16);
        v += __shfl_xor(v, 4, 16);
        v += __shfl_xor(v, 8, 16);
        acc[r] = v;
    }
    if ((lane & 15) == 0) {
        #pragma unroll
        for (int r = 0; r < 8; ++r)
            rowsum_part[(size_t)split * TWO_N + m_base + r] = acc[r];
    }
}

// ---------------------------------------------------------------------------
// Kernel 3: loss = mean_r [ -pos[r]/T + log(sum_splits rowsum_part) ]
// ---------------------------------------------------------------------------
__global__ void __launch_bounds__(256)
simclr_finalize(const float* __restrict__ pos,
                const float* __restrict__ rowsum_part,
                float* __restrict__ out)
{
    const int t = threadIdx.x;
    float local = 0.f;
    for (int r = t; r < TWO_N; r += 256) {
        float denom = 0.f;
        #pragma unroll
        for (int s = 0; s < COL_SPLITS; ++s)
            denom += rowsum_part[(size_t)s * TWO_N + r];
        local += logf(denom) - pos[r] * INV_T;
    }
    #pragma unroll
    for (int off = 16; off > 0; off >>= 1)
        local += __shfl_down(local, off, 32);
    __shared__ float red[8];
    if ((t & 31) == 0) red[t >> 5] = local;
    __syncthreads();
    if (t == 0) {
        float s = 0.f;
        #pragma unroll
        for (int w = 0; w < 8; ++w) s += red[w];
        out[0] = s / (float)TWO_N;
    }
}

// ---------------------------------------------------------------------------
extern "C" void kernel_launch(void* const* d_in, const int* in_sizes, int n_in,
                              void* d_out, int out_size, void* d_ws, size_t ws_size,
                              hipStream_t stream)
{
    (void)in_sizes; (void)n_in; (void)out_size; (void)ws_size;
    const float* ei = (const float*)d_in[0];
    const float* ej = (const float*)d_in[1];
    float* out = (float*)d_out;

    char* ws = (char*)d_ws;
    _Float16* z      = (_Float16*)ws;                                  // 4 MB
    float*    pos    = (float*)(ws + (size_t)TWO_N * DK * 2);          // 32 KB
    float*    rowsum = (float*)(ws + (size_t)TWO_N * DK * 2
                                   + (size_t)TWO_N * 4);               // 128 KB

    simclr_normalize<<<N_ROWS, 256, 0, stream>>>(ei, ej, z, pos);

    dim3 grid(TWO_N / ROWS_PER_BLK, COL_SPLITS);                       // 64 x 4
    simclr_expsum<<<grid, 256, 2 * COL_TILE * DK * 2, stream>>>(z, rowsum);

    simclr_finalize<<<1, 256, 0, stream>>>(pos, rowsum, out);
}
